// EGNN_41558103556124
// MI455X (gfx1250) — compile-verified, matrix operand order checked by reading, not run on hardware
//
#include <hip/hip_runtime.h>
#include <hip/hip_bf16.h>

// ---------------------------------------------------------------------------
// EGNN forward for MI455X (gfx1250):
//  - all GEMMs via v_wmma_f32_16x16x32_f16 (f16 in, f32 accumulate)
//  - h-row gathers staged with global_load_async_to_lds_b128 (+s_wait_asynccnt)
//  - segment sums via global_atomic_add_f32
//  - msg kernel processes 32 edges/block (2 M-tiles) so every B fragment is
//    reused twice -> halves L2 weight-fragment traffic (the dominant stream)
// ---------------------------------------------------------------------------

typedef __attribute__((ext_vector_type(16))) _Float16 v16h;
typedef __attribute__((ext_vector_type(8)))  float    v8f;
typedef __attribute__((ext_vector_type(8)))  unsigned int v8u;

#define F_    128
#define FM_   16
#define HDIM  128
#define TOUT  12
#define DIN   257
#define DH    514
#define UIN   144
#define UH    256

#define KP_MSG1 288
#define KC_MSG1 9
#define NT_MSG1 33      // 514 -> 33 tiles (528)
#define KP_MSG2 544
#define KC_MSG2 17
#define KP_UP1  160
#define KC_UP1  5
#define NT_UP1  16
#define KC_UP2  8
#define NT_UP2  8
#define KC_P    4       // K = 128
#define NT_P    8       // N = 128
#define NT_R2   1       // N = 12 (padded 16)

__device__ __forceinline__ float swishf(float x) {
  return x / (1.0f + __expf(-x));
}

// LDS byte offset of a pointer into __shared__ memory (addrspace(3) value).
__device__ __forceinline__ unsigned lds_off_of(const void* p) {
  return (unsigned)(unsigned long long)(const __attribute__((address_space(3))) char*)p;
}

// Per-lane async copy of 16 bytes global -> LDS (GLOBAL_LOAD_ASYNC_TO_LDS_B128,
// GV mode: VDST = LDS byte offset, VADDR = 64-bit global address). ASYNCcnt.
__device__ __forceinline__ void async_copy_b128(unsigned lds_off, const void* gptr) {
  asm volatile("global_load_async_to_lds_b128 %0, %1, off"
               :: "v"(lds_off), "v"((unsigned long long)gptr)
               : "memory");
}
__device__ __forceinline__ void wait_async() {
  asm volatile("s_wait_asynccnt 0x0" ::: "memory");
}

// --- A-fragment from LDS, CDNA5 16-bit A 16x32 layout -----------------------
__device__ __forceinline__ v16h load_a_frag(const _Float16* smem, int strideHalf,
                                            int lane, int kc) {
  int row = lane & 15;
  int hf  = (lane >> 4) & 1;
  const _Float16* base = smem + row * strideHalf + kc * 32 + 8 * hf;
  v8u tmp;
#pragma unroll
  for (int j = 0; j < 8; ++j) {
    int off = (j < 4) ? (2 * j) : (2 * j + 8);   // even halfword -> dword aligned
    tmp[j] = *reinterpret_cast<const unsigned int*>(base + off);
  }
  return __builtin_bit_cast(v16h, tmp);
}

// --- B-fragment from packed global: one contiguous 32B load per lane --------
__device__ __forceinline__ v16h load_b_frag(const _Float16* packed, int ntiles,
                                            int kc, int nt, int lane) {
  const _Float16* p = packed + (((size_t)(kc * ntiles + nt)) * 32 + lane) * 16;
  return *reinterpret_cast<const v16h*>(p);
}

__device__ __forceinline__ v8f wmma_f16(v16h a, v16h b, v8f c) {
  return __builtin_amdgcn_wmma_f32_16x16x32_f16(false, a, false, b,
                                                (short)0, c, false, false);
}

// ---------------------------------------------------------------------------
// Weight packing: fp32 [K x N] row-major -> f16 fragment layout
// ---------------------------------------------------------------------------
__global__ void pack_weights(const float* __restrict__ W, _Float16* __restrict__ out,
                             int K, int Ncols, int KC, int NT) {
  long t = (long)blockIdx.x * blockDim.x + threadIdx.x;
  long total = (long)KC * NT * 512;
  if (t >= total) return;
  int e    = (int)(t & 15);
  int lane = (int)((t >> 4) & 31);
  long frag = t >> 9;
  int nt = (int)(frag % NT);
  int kc = (int)(frag / NT);
  int hf  = lane >> 4;
  int col = nt * 16 + (lane & 15);
  int j = e >> 1, hi = e & 1;
  int k = kc * 32 + ((j < 4) ? 2 * j : 16 + 2 * (j - 4)) + 8 * hf + hi;
  float v = (k < K && col < Ncols) ? W[(size_t)k * Ncols + col] : 0.0f;
  out[frag * 512 + (long)lane * 16 + e] = (_Float16)v;
}

__global__ void zero_f32(float* __restrict__ p, long n) {
  long t = (long)blockIdx.x * blockDim.x + threadIdx.x;
  if (t < n) p[t] = 0.0f;
}

// h[n,f] = emb[map(z[n]), f];  also writes the f16 mirror h16
__global__ void embed_kernel(const int* __restrict__ z, const float* __restrict__ emb,
                             float* __restrict__ h, _Float16* __restrict__ h16, int n) {
  long t = (long)blockIdx.x * blockDim.x + threadIdx.x;
  if (t >= (long)n * F_) return;
  int node = (int)(t / F_), f = (int)(t % F_);
  int zz = z[node];
  int idx = (zz == 1) ? 0 : (zz - 5);
  float v = emb[(size_t)idx * F_ + f];
  h[t] = v;
  h16[t] = (_Float16)v;
}

__global__ void norm2_kernel(const float* __restrict__ pos, const int* __restrict__ ei,
                             float* __restrict__ n2, int E) {
  int t = blockIdx.x * blockDim.x + threadIdx.x;
  if (t >= E) return;
  int s = ei[t], d = ei[E + t];
  float dx = pos[(size_t)s * 3 + 0] - pos[(size_t)d * 3 + 0];
  float dy = pos[(size_t)s * 3 + 1] - pos[(size_t)d * 3 + 1];
  float dz = pos[(size_t)s * 3 + 2] - pos[(size_t)d * 3 + 2];
  n2[t] = dx * dx + dy * dy + dz * dz;
}

// ---------------------------------------------------------------------------
// Edge message kernel: 32 edges / block (2 M-tiles), 8 waves.
//   inp = [h16[dst] | h16[src] | norm2]  (32 x 288 f16, async-staged)
//   GEMM1: 32x288 @ 288x528 (each B fragment reused for both M-tiles), swish
//   GEMM2: 32x544 @ 544x16  (waves 0-3 -> tile0, 4-7 -> tile1, 4-way split-K)
//   scatter: atomicAdd into agg[dst, 0..15]
// ---------------------------------------------------------------------------
__launch_bounds__(256)
__global__ void msg_kernel(const _Float16* __restrict__ h16, const float* __restrict__ n2,
                           const int* __restrict__ ei,
                           const _Float16* __restrict__ W1p, const float* __restrict__ b1,
                           const _Float16* __restrict__ W2p, const float* __restrict__ b2,
                           float* __restrict__ agg, int E) {
  __shared__ _Float16 inp[32 * KP_MSG1];     // 18432 B
  __shared__ _Float16 hid[32 * KP_MSG2];     // 34816 B
  __shared__ float    scratch[8 * 256];      //  8192 B
  __shared__ int dstId[32];
  __shared__ int srcId[32];

  int tid  = threadIdx.x;
  int lane = tid & 31;
  int wave = tid >> 5;
  int ebase = blockIdx.x * 32;

  if (tid < 32) {
    int e = ebase + tid;
    dstId[tid] = (e < E) ? ei[E + e] : 0;
  } else if (tid < 64) {
    int r = tid - 32;
    int e = ebase + r;
    srcId[r] = (e < E) ? ei[e] : 0;
  }
  __syncthreads();

  // async-stage 64 h16 rows (32 dst + 32 src), 16 lanes x 16B per row
  {
    int lane16 = tid & 15;
    int jb0 = tid >> 4;                      // 0..15
#pragma unroll
    for (int part = 0; part < 4; ++part) {
      int job = jb0 + part * 16;             // 0..63
      int row = job & 31;
      int isSrc = job >> 5;
      int nid = isSrc ? srcId[row] : dstId[row];
      const _Float16* gp = h16 + (size_t)nid * F_ + lane16 * 8;
      unsigned loff = lds_off_of(&inp[row * KP_MSG1 + (isSrc ? F_ : 0) + lane16 * 8]);
      async_copy_b128(loff, gp);
    }
  }
  // norm2 column + K-padding (cols 256..287)
  for (int idx = tid; idx < 32 * 32; idx += 256) {
    int row = idx >> 5, c = 256 + (idx & 31);
    float v = 0.0f;
    int e = ebase + row;
    if (c == 256 && e < E) v = n2[e];
    inp[row * KP_MSG1 + c] = (_Float16)v;
  }
  // zero hid K-padding cols 528..543 (cols 514..527 compute to 0)
  for (int idx = tid; idx < 32 * 16; idx += 256) {
    int row = idx >> 4, c = 528 + (idx & 15);
    hid[row * KP_MSG2 + c] = (_Float16)0.0f;
  }
  wait_async();
  __syncthreads();

  // GEMM1 + swish -> hid (two M-tiles share each B fragment)
  for (int nt = wave; nt < NT_MSG1; nt += 8) {
    int col = nt * 16 + (lane & 15);
    float bv = (col < DH) ? b1[col] : 0.0f;
    v8f c0, c1;
#pragma unroll
    for (int i = 0; i < 8; ++i) { c0[i] = bv; c1[i] = bv; }
#pragma unroll
    for (int kc = 0; kc < KC_MSG1; ++kc) {
      v16h b  = load_b_frag(W1p, NT_MSG1, kc, nt, lane);
      v16h a0 = load_a_frag(inp, KP_MSG1, lane, kc);
      v16h a1 = load_a_frag(inp + 16 * KP_MSG1, KP_MSG1, lane, kc);
      c0 = wmma_f16(a0, b, c0);
      c1 = wmma_f16(a1, b, c1);
    }
    int row0 = 8 * (lane >> 4);
#pragma unroll
    for (int i = 0; i < 8; ++i) {
      hid[(row0 + i) * KP_MSG2 + col]        = (_Float16)swishf(c0[i]);
      hid[(16 + row0 + i) * KP_MSG2 + col]   = (_Float16)swishf(c1[i]);
    }
  }
  __syncthreads();

  // GEMM2: wave group (0-3 / 4-7) -> M-tile, 4-way split-K within group
  {
    int g = wave >> 2;
    v8f c2;
#pragma unroll
    for (int i = 0; i < 8; ++i) c2[i] = 0.0f;
    for (int kc = (wave & 3); kc < KC_MSG2; kc += 4) {
      v16h a = load_a_frag(hid + g * 16 * KP_MSG2, KP_MSG2, lane, kc);
      v16h b = load_b_frag(W2p, 1, kc, 0, lane);
      c2 = wmma_f16(a, b, c2);
    }
    int col = lane & 15;
    int row0 = 8 * (lane >> 4);
    float* sc = scratch + wave * 256;
#pragma unroll
    for (int i = 0; i < 8; ++i) sc[(row0 + i) * 16 + col] = c2[i];
  }
  __syncthreads();

  // reduce partials, swish, scatter-add (512 outputs, 2 per thread)
  for (int t2 = tid; t2 < 512; t2 += 256) {
    int row = t2 >> 4, col = t2 & 15;
    int g = row >> 4, lr = row & 15;
    float s = b2[col];
#pragma unroll
    for (int w = 0; w < 4; ++w) s += scratch[(4 * g + w) * 256 + lr * 16 + col];
    float m = swishf(s);
    int e = ebase + row;
    if (e < E) atomicAdd(&agg[(size_t)dstId[row] * FM_ + col], m);
  }
}

// ---------------------------------------------------------------------------
// Node update: u = swish([agg|h] @ Wu1 + b1);  h += u @ Wu2 + b2  (h16 mirror)
// ---------------------------------------------------------------------------
__launch_bounds__(256)
__global__ void update_kernel(float* __restrict__ h, _Float16* __restrict__ h16,
                              const float* __restrict__ agg,
                              const _Float16* __restrict__ W1p, const float* __restrict__ b1,
                              const _Float16* __restrict__ W2p, const float* __restrict__ b2,
                              int N) {
  __shared__ _Float16 uin[16 * KP_UP1];
  __shared__ _Float16 hid[16 * UH];
  int tid = threadIdx.x, lane = tid & 31, wave = tid >> 5;
  int nbase = blockIdx.x * 16;

  // agg cols 0..15 (f32->f16) + zero pad cols 144..159; one element each
  {
    int row = tid >> 4, c = tid & 15;
    int node = nbase + row;
    uin[row * KP_UP1 + c] = (_Float16)((node < N) ? agg[(size_t)node * FM_ + c] : 0.0f);
    uin[row * KP_UP1 + 144 + c] = (_Float16)0.0f;
  }
  // async-stage h16 rows into cols 16..143
  {
    int lane16 = tid & 15, row = tid >> 4;
    int node = nbase + row;
    int nid = (node < N) ? node : 0;
    async_copy_b128(lds_off_of(&uin[row * KP_UP1 + FM_ + lane16 * 8]),
                    h16 + (size_t)nid * F_ + lane16 * 8);
  }
  wait_async();
  __syncthreads();

  for (int nt = wave; nt < NT_UP1; nt += 8) {
    int col = nt * 16 + (lane & 15);
    float bv = b1[col];
    v8f c;
#pragma unroll
    for (int i = 0; i < 8; ++i) c[i] = bv;
#pragma unroll
    for (int kc = 0; kc < KC_UP1; ++kc) {
      v16h a = load_a_frag(uin, KP_UP1, lane, kc);
      v16h b = load_b_frag(W1p, NT_UP1, kc, nt, lane);
      c = wmma_f16(a, b, c);
    }
    int row0 = 8 * (lane >> 4);
#pragma unroll
    for (int i = 0; i < 8; ++i)
      hid[(row0 + i) * UH + col] = (_Float16)swishf(c[i]);
  }
  __syncthreads();

  for (int nt = wave; nt < NT_UP2; nt += 8) {
    int col = nt * 16 + (lane & 15);
    float bv = b2[col];
    v8f c;
#pragma unroll
    for (int i = 0; i < 8; ++i) c[i] = bv;
#pragma unroll
    for (int kc = 0; kc < KC_UP2; ++kc) {
      v16h a = load_a_frag(hid, UH, lane, kc);
      v16h b = load_b_frag(W2p, NT_UP2, kc, nt, lane);
      c = wmma_f16(a, b, c);
    }
    int row0 = 8 * (lane >> 4);
#pragma unroll
    for (int i = 0; i < 8; ++i) {
      int node = nbase + row0 + i;
      if (node < N) {
        size_t o = (size_t)node * F_ + col;
        float nv = c[i] + h[o];              // residual in fp32
        h[o] = nv;
        h16[o] = (_Float16)nv;
      }
    }
  }
}

// ---------------------------------------------------------------------------
// Projection + global_add_pool
// ---------------------------------------------------------------------------
__launch_bounds__(256)
__global__ void proj_pool_kernel(const _Float16* __restrict__ h16, const int* __restrict__ batch,
                                 const _Float16* __restrict__ W1p, const float* __restrict__ b1,
                                 const _Float16* __restrict__ W2p, const float* __restrict__ b2,
                                 float* __restrict__ pooled, int N) {
  __shared__ _Float16 hin[16 * F_];
  __shared__ _Float16 hid[16 * HDIM];
  __shared__ int bId[16];
  int tid = threadIdx.x, lane = tid & 31, wave = tid >> 5;
  int nbase = blockIdx.x * 16;

  if (tid < 16) {
    int node = nbase + tid;
    bId[tid] = (node < N) ? batch[node] : 0;
  }
  {
    int lane16 = tid & 15, row = tid >> 4;
    int node = nbase + row;
    int nid = (node < N) ? node : 0;
    async_copy_b128(lds_off_of(&hin[row * F_ + lane16 * 8]),
                    h16 + (size_t)nid * F_ + lane16 * 8);
  }
  wait_async();
  __syncthreads();

  for (int nt = wave; nt < NT_P; nt += 8) {
    int col = nt * 16 + (lane & 15);
    float bv = b1[col];
    v8f c;
#pragma unroll
    for (int i = 0; i < 8; ++i) c[i] = bv;
#pragma unroll
    for (int kc = 0; kc < KC_P; ++kc) {
      v16h a = load_a_frag(hin, F_, lane, kc);
      v16h b = load_b_frag(W1p, NT_P, kc, nt, lane);
      c = wmma_f16(a, b, c);
    }
    int row0 = 8 * (lane >> 4);
#pragma unroll
    for (int i = 0; i < 8; ++i)
      hid[(row0 + i) * HDIM + col] = (_Float16)swishf(c[i]);
  }
  __syncthreads();

  for (int nt = wave; nt < NT_P; nt += 8) {
    int col = nt * 16 + (lane & 15);
    float bv = b2[col];
    v8f c;
#pragma unroll
    for (int i = 0; i < 8; ++i) c[i] = bv;
#pragma unroll
    for (int kc = 0; kc < KC_P; ++kc) {
      v16h a = load_a_frag(hid, HDIM, lane, kc);
      v16h b = load_b_frag(W2p, NT_P, kc, nt, lane);
      c = wmma_f16(a, b, c);
    }
    int row0 = 8 * (lane >> 4);
#pragma unroll
    for (int i = 0; i < 8; ++i) {
      int node = nbase + row0 + i;
      if (node < N)
        atomicAdd(&pooled[(size_t)bId[row0 + i] * HDIM + col], c[i]);
    }
  }
}

// ---------------------------------------------------------------------------
// Readout: out = swish(pooled@Wr1+br1)@Wr2+br2  [G x 12]
// ---------------------------------------------------------------------------
__launch_bounds__(256)
__global__ void readout_kernel(const float* __restrict__ pooled,
                               const _Float16* __restrict__ W1p, const float* __restrict__ b1,
                               const _Float16* __restrict__ W2p, const float* __restrict__ b2,
                               float* __restrict__ out, int G) {
  __shared__ _Float16 pin[16 * HDIM];
  __shared__ _Float16 hid[16 * HDIM];
  int tid = threadIdx.x, lane = tid & 31, wave = tid >> 5;
  int gbase = blockIdx.x * 16;

  for (int idx = tid; idx < 16 * HDIM; idx += 256) {
    int row = idx / HDIM, c = idx % HDIM;
    int g = gbase + row;
    pin[idx] = (_Float16)((g < G) ? pooled[(size_t)g * HDIM + c] : 0.0f);
  }
  __syncthreads();

  for (int nt = wave; nt < NT_P; nt += 8) {
    int col = nt * 16 + (lane & 15);
    float bv = b1[col];
    v8f c;
#pragma unroll
    for (int i = 0; i < 8; ++i) c[i] = bv;
#pragma unroll
    for (int kc = 0; kc < KC_P; ++kc) {
      v16h a = load_a_frag(pin, HDIM, lane, kc);
      v16h b = load_b_frag(W1p, NT_P, kc, nt, lane);
      c = wmma_f16(a, b, c);
    }
    int row0 = 8 * (lane >> 4);
#pragma unroll
    for (int i = 0; i < 8; ++i)
      hid[(row0 + i) * HDIM + col] = (_Float16)swishf(c[i]);
  }
  __syncthreads();

  if (wave == 0) {
    int col = lane & 15;
    float bv = (col < TOUT) ? b2[col] : 0.0f;
    v8f c;
#pragma unroll
    for (int i = 0; i < 8; ++i) c[i] = bv;
#pragma unroll
    for (int kc = 0; kc < KC_P; ++kc) {
      v16h a = load_a_frag(hid, HDIM, lane, kc);
      v16h b = load_b_frag(W2p, NT_R2, kc, 0, lane);
      c = wmma_f16(a, b, c);
    }
    int row0 = 8 * (lane >> 4);
#pragma unroll
    for (int i = 0; i < 8; ++i) {
      int g = gbase + row0 + i;
      if (g < G && col < TOUT) out[(size_t)g * TOUT + col] = c[i];
    }
  }
}

// ---------------------------------------------------------------------------
extern "C" void kernel_launch(void* const* d_in, const int* in_sizes, int n_in,
                              void* d_out, int out_size, void* d_ws, size_t ws_size,
                              hipStream_t stream) {
  const float* pos    = (const float*)d_in[0];
  const int*   z      = (const int*)d_in[1];
  const int*   ei     = (const int*)d_in[2];
  const int*   batch  = (const int*)d_in[3];
  const float* emb    = (const float*)d_in[4];
  const float* W_msg1 = (const float*)d_in[5];
  const float* b_msg1 = (const float*)d_in[6];
  const float* W_msg2 = (const float*)d_in[7];
  const float* b_msg2 = (const float*)d_in[8];
  const float* W_up1  = (const float*)d_in[9];
  const float* b_up1  = (const float*)d_in[10];
  const float* W_up2  = (const float*)d_in[11];
  const float* b_up2  = (const float*)d_in[12];
  const float* Wp1    = (const float*)d_in[13];
  const float* bp1    = (const float*)d_in[14];
  const float* Wp2    = (const float*)d_in[15];
  const float* bp2    = (const float*)d_in[16];
  const float* Wr1    = (const float*)d_in[17];
  const float* br1    = (const float*)d_in[18];
  const float* Wr2    = (const float*)d_in[19];
  const float* br2    = (const float*)d_in[20];
  float* out = (float*)d_out;

  const int N = in_sizes[0] / 3;
  const int E = in_sizes[2] / 2;
  const int G = out_size / TOUT;
  const int L = in_sizes[5] / (DIN * DH);

  // ---- workspace carve-up (256B aligned) ----
  char* wp = (char*)d_ws;
  auto carve = [&](size_t bytes) {
    char* p = wp;
    wp += (bytes + 255) & ~(size_t)255;
    return p;
  };
  float*    h      = (float*)carve((size_t)N * F_ * 4);
  _Float16* h16    = (_Float16*)carve((size_t)N * F_ * 2);
  float*    n2     = (float*)carve((size_t)E * 4);
  float*    agg    = (float*)carve((size_t)N * FM_ * 4);
  float*    pooled = (float*)carve((size_t)G * HDIM * 4);
  const size_t SZ_M1 = (size_t)KC_MSG1 * NT_MSG1 * 512;
  const size_t SZ_M2 = (size_t)KC_MSG2 * 1 * 512;
  const size_t SZ_U1 = (size_t)KC_UP1 * NT_UP1 * 512;
  const size_t SZ_U2 = (size_t)KC_UP2 * NT_UP2 * 512;
  const size_t SZ_P  = (size_t)KC_P * NT_P * 512;
  const size_t SZ_R2 = (size_t)KC_P * NT_R2 * 512;
  _Float16* Wm1p = (_Float16*)carve((size_t)L * SZ_M1 * 2);
  _Float16* Wm2p = (_Float16*)carve((size_t)L * SZ_M2 * 2);
  _Float16* Wu1p = (_Float16*)carve((size_t)L * SZ_U1 * 2);
  _Float16* Wu2p = (_Float16*)carve((size_t)L * SZ_U2 * 2);
  _Float16* Wp1p = (_Float16*)carve(SZ_P * 2);
  _Float16* Wp2p = (_Float16*)carve(SZ_P * 2);
  _Float16* Wr1p = (_Float16*)carve(SZ_P * 2);
  _Float16* Wr2p = (_Float16*)carve(SZ_R2 * 2);

  auto packGrid = [](size_t total) { return (unsigned)((total + 255) / 256); };

  for (int l = 0; l < L; ++l) {
    pack_weights<<<packGrid(SZ_M1), 256, 0, stream>>>(
        W_msg1 + (size_t)l * DIN * DH, Wm1p + (size_t)l * SZ_M1, DIN, DH, KC_MSG1, NT_MSG1);
    pack_weights<<<packGrid(SZ_M2), 256, 0, stream>>>(
        W_msg2 + (size_t)l * DH * FM_, Wm2p + (size_t)l * SZ_M2, DH, FM_, KC_MSG2, 1);
    pack_weights<<<packGrid(SZ_U1), 256, 0, stream>>>(
        W_up1 + (size_t)l * UIN * UH, Wu1p + (size_t)l * SZ_U1, UIN, UH, KC_UP1, NT_UP1);
    pack_weights<<<packGrid(SZ_U2), 256, 0, stream>>>(
        W_up2 + (size_t)l * UH * F_, Wu2p + (size_t)l * SZ_U2, UH, F_, KC_UP2, NT_UP2);
  }
  pack_weights<<<packGrid(SZ_P), 256, 0, stream>>>(Wp1, Wp1p, F_, HDIM, KC_P, NT_P);
  pack_weights<<<packGrid(SZ_P), 256, 0, stream>>>(Wp2, Wp2p, HDIM, HDIM, KC_P, NT_P);
  pack_weights<<<packGrid(SZ_P), 256, 0, stream>>>(Wr1, Wr1p, HDIM, HDIM, KC_P, NT_P);
  pack_weights<<<packGrid(SZ_R2), 256, 0, stream>>>(Wr2, Wr2p, HDIM, TOUT, KC_P, NT_R2);

  embed_kernel<<<(unsigned)(((size_t)N * F_ + 255) / 256), 256, 0, stream>>>(z, emb, h, h16, N);
  norm2_kernel<<<(unsigned)((E + 255) / 256), 256, 0, stream>>>(pos, ei, n2, E);
  zero_f32<<<(unsigned)(((size_t)G * HDIM + 255) / 256), 256, 0, stream>>>(pooled, (long)G * HDIM);

  const unsigned eBlocks = (unsigned)((E + 31) / 32);
  const unsigned nBlocks = (unsigned)((N + 15) / 16);
  for (int l = 0; l < L; ++l) {
    zero_f32<<<(unsigned)(((size_t)N * FM_ + 255) / 256), 256, 0, stream>>>(agg, (long)N * FM_);
    msg_kernel<<<eBlocks, 256, 0, stream>>>(
        h16, n2, ei,
        Wm1p + (size_t)l * SZ_M1, b_msg1 + (size_t)l * DH,
        Wm2p + (size_t)l * SZ_M2, b_msg2 + (size_t)l * FM_,
        agg, E);
    update_kernel<<<nBlocks, 256, 0, stream>>>(
        h, h16, agg,
        Wu1p + (size_t)l * SZ_U1, b_up1 + (size_t)l * UH,
        Wu2p + (size_t)l * SZ_U2, b_up2 + (size_t)l * F_,
        N);
  }

  proj_pool_kernel<<<nBlocks, 256, 0, stream>>>(h16, batch, Wp1p, bp1, Wp2p, bp2, pooled, N);
  readout_kernel<<<(unsigned)((G + 15) / 16), 256, 0, stream>>>(pooled, Wr1p, br1, Wr2p, br2, out, G);
}
